// ClassicalSelfAttention_65481071398785
// MI455X (gfx1250) — compile-verified, matrix operand order checked by reading, not run on hardware
//
#include <hip/hip_runtime.h>
#include <hip/hip_bf16.h>
#include <stdint.h>

typedef __attribute__((ext_vector_type(16))) __bf16 v16bf;
typedef __attribute__((ext_vector_type(8)))  float  v8f;

#define EMBED   1024
#define NTOK    65536
#define QKV_N   3072
#define TPB     32          // tokens per block
#define SMEM_BYTES (TPB * QKV_N * 2 + 8 * 256 * 4)   // 196608 + 8192 = 204800

__device__ __forceinline__ uint16_t f32_to_bf16_rne(float f) {
    union { float f; uint32_t u; } c; c.f = f;
    uint32_t u = c.u;
    u += 0x7FFFu + ((u >> 16) & 1u);
    return (uint16_t)(u >> 16);
}

union Frag { v16bf v; uint4 u[2]; };

// ---------------------------------------------------------------- conversion
__global__ void cvt_f32_bf16(const float* __restrict__ in,
                             uint16_t* __restrict__ out, int n) {
    int i = blockIdx.x * blockDim.x + threadIdx.x;
    int stride = gridDim.x * blockDim.x;
    for (; i < n; i += stride) out[i] = f32_to_bf16_rne(in[i]);
}

// ---------------------------------------------------------------- fused kernel
// Block: 256 threads (8 waves), owns 32 token rows.
//  P1: qkv(LDS) = x[t0:t0+32] @ Wqkv^T      (WMMA, weights from L2, dbl-buffered)
//  P2: per-token 16x16 head attention       (WMMA from LDS + softmax + VALU)
//  P3: d_out = attn(LDS) @ Wo^T             (WMMA, f32 epilogue, dbl-buffered)
__global__ __launch_bounds__(256)
__attribute__((amdgpu_waves_per_eu(2)))
void fused_attn(const uint16_t* __restrict__ xb,    // [NTOK][1024] bf16
                const uint16_t* __restrict__ wqkv,  // [3072][1024] bf16
                const uint16_t* __restrict__ wob,   // [1024][1024] bf16
                float* __restrict__ out)            // [NTOK][1024] f32
{
    extern __shared__ char smem[];
    uint16_t* tile   = (uint16_t*)smem;                       // [32][3072]
    float*    s_attn = (float*)(smem + TPB * QKV_N * 2);      // [8][256]

    const int lane = threadIdx.x & 31;
    const int wave = threadIdx.x >> 5;
    const int hl   = lane >> 4;
    const int l16  = lane & 15;
    const int koff = hl * 8;              // 16-bit fragment K-interleave
    const int t0   = blockIdx.x * TPB;

    // ================= Phase 1: QKV GEMM into LDS =================
    #pragma unroll 1
    for (int pass = 0; pass < 6; ++pass) {
        const int col0 = pass * 512 + wave * 64;
        v8f acc[2][4];
        #pragma unroll
        for (int i = 0; i < 2; ++i)
            #pragma unroll
            for (int j = 0; j < 4; ++j)
                #pragma unroll
                for (int r = 0; r < 8; ++r) acc[i][j][r] = 0.0f;

        auto load_set = [&](Frag* a, Frag* b, int kk) {
            #pragma unroll
            for (int i = 0; i < 2; ++i) {
                const uint16_t* p = xb + (size_t)(t0 + i * 16 + l16) * EMBED + kk + koff;
                a[i].u[0] = *(const uint4*)p;
                a[i].u[1] = *(const uint4*)(p + 16);
            }
            #pragma unroll
            for (int j = 0; j < 4; ++j) {
                const uint16_t* p = wqkv + (size_t)(col0 + j * 16 + l16) * EMBED + kk + koff;
                b[j].u[0] = *(const uint4*)p;
                b[j].u[1] = *(const uint4*)(p + 16);
            }
        };
        auto do_wmma = [&](Frag* a, Frag* b) {
            #pragma unroll
            for (int i = 0; i < 2; ++i)
                #pragma unroll
                for (int j = 0; j < 4; ++j)
                    acc[i][j] = __builtin_amdgcn_wmma_f32_16x16x32_bf16(
                        false, a[i].v, false, b[j].v, (short)0, acc[i][j], false, false);
        };

        Frag a0[2], b0[4], a1[2], b1[4];
        load_set(a0, b0, 0);
        #pragma unroll 1
        for (int k = 0; k < EMBED; k += 64) {
            load_set(a1, b1, k + 32);        // prefetch stage B
            do_wmma(a0, b0);                 // compute stage A
            if (k + 64 < EMBED) load_set(a0, b0, k + 64);  // prefetch next stage A
            do_wmma(a1, b1);                 // compute stage B
        }

        #pragma unroll
        for (int i = 0; i < 2; ++i)
            #pragma unroll
            for (int j = 0; j < 4; ++j) {
                const int c = col0 + j * 16 + l16;
                #pragma unroll
                for (int r = 0; r < 8; ++r) {
                    const int rr = i * 16 + r + hl * 8;       // C/D layout row
                    tile[rr * QKV_N + c] = f32_to_bf16_rne(acc[i][j][r]);
                }
            }
    }
    __syncthreads();

    // ================= Phase 2: head attention (wave -> 4 tokens) =================
    const float scale = 0.125f;   // 1/sqrt(64)
    #pragma unroll 1
    for (int tt = 0; tt < 4; ++tt) {
        const int tok = wave * 4 + tt;
        uint16_t* q  = tile + tok * QKV_N;          // [16][64]
        uint16_t* km = q + 1024;
        uint16_t* vm = q + 2048;

        v8f s;
        #pragma unroll
        for (int r = 0; r < 8; ++r) s[r] = 0.0f;

        #pragma unroll
        for (int kc = 0; kc < 64; kc += 32) {
            Frag a, b;
            const uint16_t* pa = q  + l16 * 64 + kc + koff;
            const uint16_t* pb = km + l16 * 64 + kc + koff;
            a.u[0] = *(const uint4*)pa; a.u[1] = *(const uint4*)(pa + 16);
            b.u[0] = *(const uint4*)pb; b.u[1] = *(const uint4*)(pb + 16);
            s = __builtin_amdgcn_wmma_f32_16x16x32_bf16(
                    false, a.v, false, b.v, (short)0, s, false, false);
        }

        // row softmax: row M = r + 8*hl lives across the 16 lanes of a half-wave
        #pragma unroll
        for (int r = 0; r < 8; ++r) {
            float x = s[r] * scale;
            float m = x;
            #pragma unroll
            for (int msk = 1; msk <= 8; msk <<= 1)
                m = fmaxf(m, __shfl_xor(m, msk, 32));
            float e = __expf(x - m);
            float sum = e;
            #pragma unroll
            for (int msk = 1; msk <= 8; msk <<= 1)
                sum += __shfl_xor(sum, msk, 32);
            s_attn[wave * 256 + (r + hl * 8) * 16 + l16] = e / sum;
        }

        // out[h][d] = sum_g attn[h][g] * v[g][d]; lane: h=l16, d in [hl*32, +32)
        float accv[32];
        #pragma unroll
        for (int d = 0; d < 32; ++d) accv[d] = 0.0f;
        const int d0 = hl * 32;
        #pragma unroll 1
        for (int g = 0; g < 16; ++g) {
            const float ag = s_attn[wave * 256 + l16 * 16 + g];
            const uint4* vq = (const uint4*)(vm + g * 64 + d0);
            #pragma unroll
            for (int q4 = 0; q4 < 4; ++q4) {
                uint4 w = vq[q4];
                const uint32_t ww[4] = {w.x, w.y, w.z, w.w};
                #pragma unroll
                for (int e2 = 0; e2 < 4; ++e2) {
                    union { uint32_t u; float f; } lo, hi;
                    lo.u = ww[e2] << 16;
                    hi.u = ww[e2] & 0xFFFF0000u;
                    accv[q4 * 8 + e2 * 2 + 0] = fmaf(ag, lo.f, accv[q4 * 8 + e2 * 2 + 0]);
                    accv[q4 * 8 + e2 * 2 + 1] = fmaf(ag, hi.f, accv[q4 * 8 + e2 * 2 + 1]);
                }
            }
        }
        // overwrite dead q region with this token's attention output
        uint16_t* op = q + l16 * 64 + d0;
        #pragma unroll
        for (int d = 0; d < 32; ++d) op[d] = f32_to_bf16_rne(accv[d]);
    }
    __syncthreads();

    // ================= Phase 3: O-projection from LDS =================
    #pragma unroll 1
    for (int pass = 0; pass < 2; ++pass) {
        const int col0 = pass * 512 + wave * 64;
        v8f acc[2][4];
        #pragma unroll
        for (int i = 0; i < 2; ++i)
            #pragma unroll
            for (int j = 0; j < 4; ++j)
                #pragma unroll
                for (int r = 0; r < 8; ++r) acc[i][j][r] = 0.0f;

        auto load_set = [&](Frag* a, Frag* b, int kk) {
            #pragma unroll
            for (int i = 0; i < 2; ++i) {
                const uint16_t* p = tile + (i * 16 + l16) * QKV_N + kk + koff;
                a[i].u[0] = *(const uint4*)p;
                a[i].u[1] = *(const uint4*)(p + 16);
            }
            #pragma unroll
            for (int j = 0; j < 4; ++j) {
                const uint16_t* p = wob + (size_t)(col0 + j * 16 + l16) * EMBED + kk + koff;
                b[j].u[0] = *(const uint4*)p;
                b[j].u[1] = *(const uint4*)(p + 16);
            }
        };
        auto do_wmma = [&](Frag* a, Frag* b) {
            #pragma unroll
            for (int i = 0; i < 2; ++i)
                #pragma unroll
                for (int j = 0; j < 4; ++j)
                    acc[i][j] = __builtin_amdgcn_wmma_f32_16x16x32_bf16(
                        false, a[i].v, false, b[j].v, (short)0, acc[i][j], false, false);
        };

        Frag a0[2], b0[4], a1[2], b1[4];
        load_set(a0, b0, 0);
        #pragma unroll 1
        for (int k = 0; k < EMBED; k += 64) {
            load_set(a1, b1, k + 32);
            do_wmma(a0, b0);
            if (k + 64 < EMBED) load_set(a0, b0, k + 64);
            do_wmma(a1, b1);
        }

        #pragma unroll
        for (int i = 0; i < 2; ++i)
            #pragma unroll
            for (int j = 0; j < 4; ++j) {
                const int c = col0 + j * 16 + l16;
                #pragma unroll
                for (int r = 0; r < 8; ++r) {
                    const int rr = t0 + i * 16 + r + hl * 8;
                    out[(size_t)rr * EMBED + c] = acc[i][j][r];
                }
            }
    }
}

// ---------------------------------------------------------------- launch
extern "C" void kernel_launch(void* const* d_in, const int* in_sizes, int n_in,
                              void* d_out, int out_size, void* d_ws, size_t ws_size,
                              hipStream_t stream) {
    (void)in_sizes; (void)n_in; (void)out_size; (void)ws_size;
    const float* x  = (const float*)d_in[0];
    const float* Wq = (const float*)d_in[1];
    const float* Wk = (const float*)d_in[2];
    const float* Wv = (const float*)d_in[3];
    const float* Wo = (const float*)d_in[4];
    float* out = (float*)d_out;

    uint8_t* ws = (uint8_t*)d_ws;
    size_t off = 0;
    uint16_t* xb   = (uint16_t*)(ws + off); off += (size_t)NTOK * EMBED * 2;       // 128 MB
    uint16_t* wqkv = (uint16_t*)(ws + off); off += (size_t)3 * EMBED * EMBED * 2;  // 6 MB
    uint16_t* wob  = (uint16_t*)(ws + off); off += (size_t)EMBED * EMBED * 2;      // 2 MB

    const int WN = EMBED * EMBED;
    cvt_f32_bf16<<<4096, 256, 0, stream>>>(x,  xb,            NTOK * EMBED);
    cvt_f32_bf16<<<1024, 256, 0, stream>>>(Wq, wqkv,          WN);
    cvt_f32_bf16<<<1024, 256, 0, stream>>>(Wk, wqkv + WN,     WN);
    cvt_f32_bf16<<<1024, 256, 0, stream>>>(Wv, wqkv + 2 * WN, WN);
    cvt_f32_bf16<<<1024, 256, 0, stream>>>(Wo, wob,           WN);

    fused_attn<<<NTOK / TPB, 256, SMEM_BYTES, stream>>>(xb, wqkv, wob, out);
}
